// TiReMGE_45440753991796
// MI455X (gfx1250) — matrix operand order repaired and checked by reading, not compiled, not run on hardware
//
#include <hip/hip_runtime.h>

// ---------------------------------------------------------------------------
// GCN pipeline for MI455X (gfx1250, wave32, WMMA).
//  - Weights pre-packed once per launch into contiguous-k WMMA B-fragment layout
//    (f16, 32 contiguous bytes per lane per fragment -> global_load_b128).
//  - Activations pre-converted to f16 row-major (A fragments = 2x v8h loads).
//  - GEMMs: v_wmma_f32_16x16x32_f16, fp32 accumulate, per-wave 16x(NB*16) tile.
//  - Edge aggregation: coalesced gathers + global_atomic_add_f32 scatters;
//    self-loop term folded analytically into the epilogue (t[i]*dinv[i]^2).
// ---------------------------------------------------------------------------

typedef __attribute__((ext_vector_type(16))) _Float16 v16h;
typedef __attribute__((ext_vector_type(8)))  _Float16 v8h;
typedef __attribute__((ext_vector_type(8)))  float    v8f;

#define NODES 50000
#define FEAT  512

// ---------------------------------------------------------------------------
// WMMA GEMM: C[M,N] = A[M,K] @ B[K,N] (+ bias), A f16 row-major, Bp packed.
// Bp layout: Bp[((nt*KT + kt)*32 + lane)*16 + j]
//   lane<16 : holds column n = nt*16+lane,  j -> K = kt*32 + j        (0..15)
//   lane>=16: holds column n = nt*16+lane-16, j -> K = kt*32 + 16 + j (16..31)
// Block = 128 threads = 4 waves; wave owns one 16-row M tile, NB N-subtiles.
// ---------------------------------------------------------------------------
template <int NB>
__global__ void __launch_bounds__(128)
wmma_gemm_kernel(const _Float16* __restrict__ A, const _Float16* __restrict__ Bp,
                 const float* __restrict__ bias, float* __restrict__ C,
                 int M, int N, int K)
{
    const int lane  = threadIdx.x & 31;
    const int wave  = threadIdx.x >> 5;
    const int mtile = blockIdx.x * 4 + wave;
    if (mtile * 16 >= M) return;                 // wave-uniform: EXEC stays full

    const int nt0   = blockIdx.y * NB;           // first 16-wide N subtile
    const int row   = mtile * 16 + (lane & 15);  // A row owned by this lane
    const int khalf = lane >> 4;                 // 0 or 1 (lane group)
    const int KT    = K >> 5;                    // 32-wide K steps

    v8f acc[NB];
#pragma unroll
    for (int nb = 0; nb < NB; ++nb) acc[nb] = (v8f){};

    const _Float16* arow = A + (long long)row * K + khalf * 16;

    for (int kt = 0; kt < KT; ++kt) {
        // ---- A fragment: 16x32 f16.  Lane holds M=row.
        //      v[0..7]  = K = kt*32 + khalf*16 + {0..7}
        //      v[8..15] = K = kt*32 + khalf*16 + 8 + {0..7}
        const _Float16* ap = arow + kt * 32;
        __builtin_prefetch(ap + 64, 0, 1);       // global_prefetch next k-steps
        const v8h lo = *reinterpret_cast<const v8h*>(ap);
        const v8h hi = *reinterpret_cast<const v8h*>(ap + 8);
        v16h af;
#pragma unroll
        for (int j = 0; j < 8; ++j) { af[j] = lo[j]; af[8 + j] = hi[j]; }

#pragma unroll
        for (int nb = 0; nb < NB; ++nb) {
            // ---- B fragment: one contiguous 32B v16h load per lane.
            const v16h bf = *reinterpret_cast<const v16h*>(
                Bp + (((long long)(nt0 + nb) * KT + kt) * 32 + lane) * 16);

            acc[nb] = __builtin_amdgcn_wmma_f32_16x16x32_f16(
                /*neg_a=*/false, af, /*neg_b=*/false, bf,
                /*c_mod=*/(short)0, acc[nb],
                /*reuse_a=*/false, /*reuse_b=*/false);
        }
    }

    // ---- store: C/D 16x16 f32, VGPR r -> M = r + 8*khalf, N = lane&15
#pragma unroll
    for (int nb = 0; nb < NB; ++nb) {
        const int cn = (nt0 + nb) * 16 + (lane & 15);
        const float bia = bias ? bias[cn] : 0.0f;
#pragma unroll
        for (int r = 0; r < 8; ++r) {
            const int cm = mtile * 16 + r + khalf * 8;
            C[(long long)cm * N + cn] = acc[nb][r] + bia;
        }
    }
}

// ---------------------------------------------------------------------------
// Pack fp32 B[K,N] -> f16 fragment layout described above.
// One thread per packed element (total = N*K).
// ---------------------------------------------------------------------------
__global__ void pack_b_kernel(const float* __restrict__ B, _Float16* __restrict__ Bp,
                              int N, int K)
{
    const int KT = K >> 5;
    long long id = (long long)blockIdx.x * blockDim.x + threadIdx.x;
    if (id >= (long long)N * K) return;
    const int j    = (int)(id & 15);
    const int lane = (int)((id >> 4) & 31);
    const long long t = id >> 9;           // nt*KT + kt
    const int kt = (int)(t % KT);
    const int nt = (int)(t / KT);
    const int ksrc = kt * 32 + (lane >> 4) * 16 + j;
    const int n    = nt * 16 + (lane & 15);
    Bp[id] = (_Float16)B[(long long)ksrc * N + n];
}

__global__ void cvt_f16_kernel(const float* __restrict__ s, _Float16* __restrict__ d,
                               long long n)
{
    long long i = (long long)blockIdx.x * blockDim.x + threadIdx.x;
    if (i < n) d[i] = (_Float16)s[i];
}

// ---------------------------------------------------------------------------
// Helpers: zero, degree init/accumulate/rsqrt, edge scatter, epilogue.
// ---------------------------------------------------------------------------
__global__ void zero_kernel(float* __restrict__ p, long long n)
{
    long long i = (long long)blockIdx.x * blockDim.x + threadIdx.x;
    if (i < n) p[i] = 0.0f;
}

__global__ void deg_init_kernel(float* __restrict__ d1, float* __restrict__ d2, int n)
{
    int i = blockIdx.x * blockDim.x + threadIdx.x;
    if (i < n) { d1[i] = 1.0f; d2[i] = 1.0f; }   // self-loop contributes 1
}

__global__ void deg_accum_kernel(const int* __restrict__ dst, float* __restrict__ deg, int E)
{
    int e = blockIdx.x * blockDim.x + threadIdx.x;
    if (e < E) atomicAdd(&deg[dst[e]], 1.0f);
}

__global__ void make_dinv_kernel(float* __restrict__ deg, int n)
{
    int i = blockIdx.x * blockDim.x + threadIdx.x;
    if (i < n) deg[i] = rsqrtf(fmaxf(deg[i], 1e-12f));
}

// out[dst, f] += h[src, f] * dinv[src] * dinv[dst]   (edges only)
__global__ void edge_agg_kernel(const float* __restrict__ h, const int* __restrict__ ei,
                                const float* __restrict__ dinv, float* __restrict__ out,
                                int E, int dshift)
{
    long long id = (long long)blockIdx.x * blockDim.x + threadIdx.x;
    if (id >= ((long long)E << dshift)) return;
    const int e = (int)(id >> dshift);
    const int f = (int)(id & ((1 << dshift) - 1));
    const int s = ei[e];         // src row
    const int d = ei[E + e];     // dst row
    const float v = h[((long long)s << dshift) + f] * dinv[s] * dinv[d];
    atomicAdd(&out[((long long)d << dshift) + f], v);
}

// out = [relu]( agg + t*dinv^2 + bias ) [+ addsrc]     (out may alias t)
__global__ void epilogue_kernel(const float* __restrict__ agg, const float* __restrict__ t,
                                const float* __restrict__ dinv, const float* __restrict__ bias,
                                const float* __restrict__ addsrc, float* __restrict__ out,
                                int n, int dshift, int relu)
{
    long long id = (long long)blockIdx.x * blockDim.x + threadIdx.x;
    if (id >= ((long long)n << dshift)) return;
    const int i = (int)(id >> dshift);
    const int f = (int)(id & ((1 << dshift) - 1));
    const float di = dinv[i];
    float v = agg[id] + t[id] * di * di + bias[f];
    if (relu) v = fmaxf(v, 0.0f);
    if (addsrc) v += addsrc[id];
    out[id] = v;
}

// ---------------------------------------------------------------------------
// Launch
// ---------------------------------------------------------------------------
extern "C" void kernel_launch(void* const* d_in, const int* in_sizes, int n_in,
                              void* d_out, int out_size, void* d_ws, size_t ws_size,
                              hipStream_t stream)
{
    (void)n_in; (void)out_size; (void)ws_size;

    const float* x   = (const float*)d_in[0];
    const int*   e1  = (const int*)d_in[1];
    const int*   e2  = (const int*)d_in[2];
    const float* W10 = (const float*)d_in[3];
    const float* b10 = (const float*)d_in[4];
    const float* W20 = (const float*)d_in[5];
    const float* b20 = (const float*)d_in[6];
    const float* W21 = (const float*)d_in[7];
    const float* b21 = (const float*)d_in[8];
    const float* Wfc = (const float*)d_in[9];
    const float* bfc = (const float*)d_in[10];

    const int N  = NODES;
    const int E1 = in_sizes[1] / 2;
    const int E2 = in_sizes[2] / 2;

    // ---- workspace layout: fp32 region, then f16 region
    float* wf    = (float*)d_ws;
    float* dinv1 = wf;                  wf += N;
    float* dinv2 = wf;                  wf += N;
    float* T     = wf;                  wf += (size_t)N * 256;  // GEMM out / hs
    float* AG    = wf;                  wf += (size_t)N * 256;  // aggregation
    float* H1    = wf;                  wf += (size_t)N * 256;  // branch-1 out
    float* H2    = wf;                  wf += (size_t)N * 64;   // layer-2 out (fp32)

    _Float16* hp   = (_Float16*)wf;
    _Float16* Xh   = hp;                hp += (size_t)N * FEAT; // x in f16
    _Float16* H2h  = hp;                hp += (size_t)N * 64;   // h2a in f16
    _Float16* HSh  = hp;                hp += (size_t)N * 256;  // hs in f16
    _Float16* W10p = hp;                hp += (size_t)FEAT * 256;
    _Float16* W20p = hp;                hp += (size_t)FEAT * 64;
    _Float16* W21p = hp;                hp += (size_t)64 * 256;
    _Float16* Wfcp = hp;                hp += (size_t)256 * 16;

    const int TB = 256;
    auto blk = [](long long n, int t) { return (unsigned)((n + t - 1) / t); };
    const unsigned GX = (unsigned)(((N + 15) / 16 + 3) / 4);    // M-tiles / 4 waves

    // ---- pack weights into WMMA fragment layout, convert x to f16
    pack_b_kernel<<<blk((long long)FEAT * 256, TB), TB, 0, stream>>>(W10, W10p, 256, FEAT);
    pack_b_kernel<<<blk((long long)FEAT * 64,  TB), TB, 0, stream>>>(W20, W20p, 64,  FEAT);
    pack_b_kernel<<<blk((long long)64 * 256,   TB), TB, 0, stream>>>(W21, W21p, 256, 64);
    pack_b_kernel<<<blk((long long)256 * 16,   TB), TB, 0, stream>>>(Wfc, Wfcp, 16,  256);
    cvt_f16_kernel<<<blk((long long)N * FEAT, TB), TB, 0, stream>>>(x, Xh, (long long)N * FEAT);

    // ---- symmetric-norm degrees (include self loop), then rsqrt
    deg_init_kernel<<<blk(N, TB), TB, 0, stream>>>(dinv1, dinv2, N);
    deg_accum_kernel<<<blk(E1, TB), TB, 0, stream>>>(e1 + E1, dinv1, E1);
    deg_accum_kernel<<<blk(E2, TB), TB, 0, stream>>>(e2 + E2, dinv2, E2);
    make_dinv_kernel<<<blk(N, TB), TB, 0, stream>>>(dinv1, N);
    make_dinv_kernel<<<blk(N, TB), TB, 0, stream>>>(dinv2, N);

    // ---- layer 1: h1 = relu( norm-agg(x@W10, e1) + b10 )
    wmma_gemm_kernel<4><<<dim3(GX, 4), 128, 0, stream>>>(Xh, W10p, nullptr, T, N, 256, FEAT);
    zero_kernel<<<blk((long long)N << 8, TB), TB, 0, stream>>>(AG, (long long)N << 8);
    edge_agg_kernel<<<blk((long long)E1 << 8, TB), TB, 0, stream>>>(T, e1, dinv1, AG, E1, 8);
    epilogue_kernel<<<blk((long long)N << 8, TB), TB, 0, stream>>>(AG, T, dinv1, b10, nullptr, H1, N, 8, 1);

    // ---- layer 2: h2a = norm-agg(x@W20, e2) + b20   (no relu, 64-wide)
    wmma_gemm_kernel<4><<<dim3(GX, 1), 128, 0, stream>>>(Xh, W20p, nullptr, T, N, 64, FEAT);
    zero_kernel<<<blk((long long)N << 6, TB), TB, 0, stream>>>(AG, (long long)N << 6);
    edge_agg_kernel<<<blk((long long)E2 << 6, TB), TB, 0, stream>>>(T, e2, dinv2, AG, E2, 6);
    epilogue_kernel<<<blk((long long)N << 6, TB), TB, 0, stream>>>(AG, T, dinv2, b20, nullptr, H2, N, 6, 0);
    cvt_f16_kernel<<<blk((long long)N * 64, TB), TB, 0, stream>>>(H2, H2h, (long long)N * 64);

    // ---- layer 3: hs = relu( norm-agg(h2a@W21, e2) + b21 ) + h1   (into T)
    wmma_gemm_kernel<4><<<dim3(GX, 4), 128, 0, stream>>>(H2h, W21p, nullptr, T, N, 256, 64);
    zero_kernel<<<blk((long long)N << 8, TB), TB, 0, stream>>>(AG, (long long)N << 8);
    edge_agg_kernel<<<blk((long long)E2 << 8, TB), TB, 0, stream>>>(T, e2, dinv2, AG, E2, 8);
    epilogue_kernel<<<blk((long long)N << 8, TB), TB, 0, stream>>>(AG, T, dinv2, b21, H1, T, N, 8, 1);
    cvt_f16_kernel<<<blk((long long)N * 256, TB), TB, 0, stream>>>(T, HSh, (long long)N * 256);

    // ---- final dense: out = hs @ Wfc + bfc   (N = 16 -> one subtile per wave)
    wmma_gemm_kernel<1><<<dim3(GX, 1), 128, 0, stream>>>(HSh, Wfcp, bfc, (float*)d_out, N, 16, 256);
}
